// CrossAttention_72146860638912
// MI455X (gfx1250) — matrix operand order repaired and checked by reading, not compiled, not run on hardware
//
#include <hip/hip_runtime.h>

// B=8, N=4096, C=1024, H=16, D=64, scale = 1/8
#define BB 8
#define NN 4096
#define CC 1024
#define HH 16
#define DD 64
#define ATTN_SCALE 0.125f
#define KP 40   // LDS row stride in ushorts for 32-wide K tiles (16B-aligned chunks, bank-spread)

typedef __attribute__((ext_vector_type(16))) __bf16          v16bf;
typedef __attribute__((ext_vector_type(8)))  unsigned short  v8u16;
typedef __attribute__((ext_vector_type(8)))  float           v8f;

union FragU { v8u16 h[2]; v16bf v; };

static __device__ __forceinline__ v16bf mkfrag(v8u16 lo, v8u16 hi) {
    FragU u; u.h[0] = lo; u.h[1] = hi; return u.v;
}

// f32 -> bf16 via native convert (backend selects v_cvt hardware op on gfx1250)
static __device__ __forceinline__ unsigned short f2bf(float f) {
    return __builtin_bit_cast(unsigned short, (__bf16)f);
}

static __device__ __forceinline__ v8f wmma_bf16(v16bf a, v16bf b, v8f c) {
    return __builtin_amdgcn_wmma_f32_16x16x32_bf16(false, a, false, b, (short)0, c, false, false);
}

// ---------------------------------------------------------------------------
// Phase 1: dst = X @ W^T  (32768x1024 x 1024x1024), X,W f32 -> bf16 via LDS.
// transposed=1: write dst as [b][h][d][n] bf16 (packed 16B stores)
// transposed=0: write dst as natural (B*N, C) bf16
// ---------------------------------------------------------------------------
__global__ __launch_bounds__(256) void qkv_gemm_kernel(
    const float* __restrict__ X, const float* __restrict__ W,
    unsigned short* __restrict__ dst, int transposed)
{
    __shared__ unsigned short Xs[128 * KP];
    __shared__ unsigned short Ws[128 * KP];

    const int tid  = threadIdx.x;
    const int m0   = blockIdx.x * 128;
    const int o0   = blockIdx.y * 128;
    const int wid  = tid >> 5;
    const int lane = tid & 31;
    const int wm   = (wid >> 2) * 64;       // wave M offset: 0 / 64
    const int wo   = (wid & 3) * 32;        // wave O offset: 0..96
    const int lrow = lane & 15;
    const int c0   = (lane < 16) ? 0 : 8;   // A-fragment k-chunk base
    const int kb   = (lane < 16) ? 0 : 16;  // unused; B fragments also use c0
    const int lmb  = (lane < 16) ? 0 : 8;   // C/D-tile M base

    v8f acc[4][2];
    #pragma unroll
    for (int i = 0; i < 4; ++i)
        #pragma unroll
        for (int j = 0; j < 2; ++j) acc[i][j] = {};

    for (int kk = 0; kk < CC / 32; ++kk) {
        const int k0 = kk * 32;
        __syncthreads();
        // cooperative load+convert: 128x32 f32 tiles of X and W -> bf16 LDS
        #pragma unroll
        for (int j = 0; j < 4; ++j) {
            const int g   = tid + j * 256;   // 1024 float4 groups
            const int row = g >> 3;          // 8 groups (32 floats) per row
            const int c4  = (g & 7) * 4;
            float4 xv = *reinterpret_cast<const float4*>(X + (size_t)(m0 + row) * CC + k0 + c4);
            Xs[row * KP + c4 + 0] = f2bf(xv.x);
            Xs[row * KP + c4 + 1] = f2bf(xv.y);
            Xs[row * KP + c4 + 2] = f2bf(xv.z);
            Xs[row * KP + c4 + 3] = f2bf(xv.w);
            float4 wv = *reinterpret_cast<const float4*>(W + (size_t)(o0 + row) * CC + k0 + c4);
            Ws[row * KP + c4 + 0] = f2bf(wv.x);
            Ws[row * KP + c4 + 1] = f2bf(wv.y);
            Ws[row * KP + c4 + 2] = f2bf(wv.z);
            Ws[row * KP + c4 + 3] = f2bf(wv.w);
            // prefetch next k-tile (pure HBM stream)
            if (kk + 1 < CC / 32) {
                __builtin_prefetch(X + (size_t)(m0 + row) * CC + k0 + 32 + c4, 0, 0);
                __builtin_prefetch(W + (size_t)(o0 + row) * CC + k0 + 32 + c4, 0, 0);
            }
        }
        __syncthreads();

        v16bf afr[4], bfr[2];
        #pragma unroll
        for (int ms = 0; ms < 4; ++ms) {
            const unsigned short* p = &Xs[(wm + ms * 16 + lrow) * KP + c0];
            afr[ms] = mkfrag(*(const v8u16*)p, *(const v8u16*)(p + 16));
        }
        #pragma unroll
        for (int os = 0; os < 2; ++os) {
            const unsigned short* p = &Ws[(wo + os * 16 + lrow) * KP + c0];
            bfr[os] = mkfrag(*(const v8u16*)p, *(const v8u16*)(p + 16));
        }
        #pragma unroll
        for (int ms = 0; ms < 4; ++ms)
            #pragma unroll
            for (int os = 0; os < 2; ++os)
                acc[ms][os] = wmma_bf16(afr[ms], bfr[os], acc[ms][os]);
    }

    if (transposed) {
        const int bIdx = m0 >> 12;  // N = 4096 rows per batch, m0 multiple of 128
        #pragma unroll
        for (int ms = 0; ms < 4; ++ms) {
            const int n0 = (m0 + wm + ms * 16 + lmb) & (NN - 1);
            #pragma unroll
            for (int os = 0; os < 2; ++os) {
                const int o = o0 + wo + os * 16 + lrow;
                const int h = o >> 6, d = o & 63;
                v8u16 pk;
                #pragma unroll
                for (int j = 0; j < 8; ++j) pk[j] = f2bf(acc[ms][os][j]);
                *reinterpret_cast<v8u16*>(dst + ((size_t)(bIdx * HH + h) * DD + d) * NN + n0) = pk;
            }
        }
    } else {
        #pragma unroll
        for (int ms = 0; ms < 4; ++ms)
            #pragma unroll
            for (int os = 0; os < 2; ++os) {
                const int o = o0 + wo + os * 16 + lrow;
                #pragma unroll
                for (int j = 0; j < 8; ++j) {
                    const size_t m = (size_t)(m0 + wm + ms * 16 + lmb + j);
                    dst[m * CC + o] = f2bf(acc[ms][os][j]);
                }
            }
    }
}

// ---------------------------------------------------------------------------
// Phase 2: per (b,h): S = scale * Q K^T (64x64 over N=4096), softmax rows,
//          out_t[n,d] = sum_e V[n,e] * P[d,e]; Ot in natural (B,N,C) bf16.
// Grid: B*H workgroups, 256 threads (8 waves).
// ---------------------------------------------------------------------------
__global__ __launch_bounds__(256) void attn_kernel(
    const unsigned short* __restrict__ Qt, const unsigned short* __restrict__ Kt,
    const unsigned short* __restrict__ Vn, unsigned short* __restrict__ Ot)
{
    __shared__ float S[DD * DD];
    __shared__ unsigned short P[DD * DD];

    const int tid  = threadIdx.x;
    const int wid  = tid >> 5;
    const int lane = tid & 31;
    const int lrow = lane & 15;
    const int c0   = (lane < 16) ? 0 : 8;
    const int kb   = (lane < 16) ? 0 : 16;
    const int lmb  = (lane < 16) ? 0 : 8;
    const int bh   = blockIdx.x;
    const int b    = bh >> 4, h = bh & 15;
    const size_t qkBase = (size_t)bh * DD * NN;

    // ---- stage 1: each wave owns 2 of the 16 S-tiles, full-N contraction ----
    const int dt  = wid & 3;
    const int et0 = (wid >> 2) * 2;
    v8f accS[2]; accS[0] = {}; accS[1] = {};
    const unsigned short* qrow  = Qt + qkBase + (size_t)(dt * 16 + lrow) * NN;
    const unsigned short* krow0 = Kt + qkBase + (size_t)((et0 + 0) * 16 + lrow) * NN;
    const unsigned short* krow1 = Kt + qkBase + (size_t)((et0 + 1) * 16 + lrow) * NN;
    for (int kk = 0; kk < NN / 32; ++kk) {
        const int n0 = kk * 32;
        v16bf a  = mkfrag(*(const v8u16*)(qrow  + n0 + c0), *(const v8u16*)(qrow  + n0 + c0 + 16));
        v16bf b0 = mkfrag(*(const v8u16*)(krow0 + n0 + c0), *(const v8u16*)(krow0 + n0 + c0 + 16));
        v16bf b1 = mkfrag(*(const v8u16*)(krow1 + n0 + c0), *(const v8u16*)(krow1 + n0 + c0 + 16));
        // prefetch next 32-column chunk of the three streaming rows
        if (kk + 1 < NN / 32) {
            __builtin_prefetch(qrow  + n0 + 32, 0, 0);
            __builtin_prefetch(krow0 + n0 + 32, 0, 0);
            __builtin_prefetch(krow1 + n0 + 32, 0, 0);
        }
        accS[0] = wmma_bf16(a, b0, accS[0]);
        accS[1] = wmma_bf16(a, b1, accS[1]);
    }
    #pragma unroll
    for (int t = 0; t < 2; ++t) {
        const int e = (et0 + t) * 16 + lrow;
        #pragma unroll
        for (int j = 0; j < 8; ++j) {
            const int d = dt * 16 + lmb + j;
            S[d * DD + e] = accS[t][j] * ATTN_SCALE;
        }
    }
    __syncthreads();

    // ---- softmax over rows (64 rows handled by threads 0..63) ----
    if (tid < DD) {
        float mx = -3.0e38f;
        for (int e = 0; e < DD; ++e) mx = fmaxf(mx, S[tid * DD + e]);
        float sm = 0.f;
        for (int e = 0; e < DD; ++e) sm += __expf(S[tid * DD + e] - mx);
        const float inv = 1.0f / sm;
        for (int e = 0; e < DD; ++e) P[tid * DD + e] = f2bf(__expf(S[tid * DD + e] - mx) * inv);
    }
    __syncthreads();

    // ---- stage 2: out_t[n,d] = sum_e V[n,e] * P[d,e] ----
    v16bf bP[4][2];  // B fragments from LDS, hoisted (same for all n-tiles)
    #pragma unroll
    for (int dt2 = 0; dt2 < 4; ++dt2)
        #pragma unroll
        for (int ks = 0; ks < 2; ++ks) {
            const unsigned short* p = &P[(dt2 * 16 + lrow) * DD + ks * 32 + c0];
            bP[dt2][ks] = mkfrag(*(const v8u16*)p, *(const v8u16*)(p + 16));
        }
    const size_t vBase = (size_t)b * NN * CC + (size_t)h * DD;
    for (int nt = 0; nt < 32; ++nt) {
        const int n0 = wid * 512 + nt * 16;
        v16bf aV[2];
        #pragma unroll
        for (int ks = 0; ks < 2; ++ks) {
            const unsigned short* p = Vn + vBase + (size_t)(n0 + lrow) * CC + ks * 32 + c0;
            aV[ks] = mkfrag(*(const v8u16*)p, *(const v8u16*)(p + 16));
        }
        if (nt + 1 < 32)
            __builtin_prefetch(Vn + vBase + (size_t)(n0 + 16 + lrow) * CC + c0, 0, 0);
        v8f accO[4];
        #pragma unroll
        for (int i = 0; i < 4; ++i) accO[i] = {};
        #pragma unroll
        for (int dt2 = 0; dt2 < 4; ++dt2)
            #pragma unroll
            for (int ks = 0; ks < 2; ++ks)
                accO[dt2] = wmma_bf16(aV[ks], bP[dt2][ks], accO[dt2]);
        #pragma unroll
        for (int dt2 = 0; dt2 < 4; ++dt2) {
            const int d = dt2 * 16 + lrow;
            #pragma unroll
            for (int j = 0; j < 8; ++j) {
                const size_t n = (size_t)(n0 + lmb + j);
                Ot[((size_t)b * NN + n) * CC + h * DD + d] = f2bf(accO[dt2][j]);
            }
        }
    }
}

// ---------------------------------------------------------------------------
// Phase 3: out = Ot(bf16) @ Wp^T + bp, f32 output.
// ---------------------------------------------------------------------------
__global__ __launch_bounds__(256) void proj_kernel(
    const unsigned short* __restrict__ A, const float* __restrict__ W,
    const float* __restrict__ bias, float* __restrict__ out)
{
    __shared__ unsigned short Ws[128 * KP];

    const int tid  = threadIdx.x;
    const int m0   = blockIdx.x * 128;
    const int o0   = blockIdx.y * 128;
    const int wid  = tid >> 5;
    const int lane = tid & 31;
    const int wm   = (wid >> 2) * 64;
    const int wo   = (wid & 3) * 32;
    const int lrow = lane & 15;
    const int c0   = (lane < 16) ? 0 : 8;
    const int kb   = (lane < 16) ? 0 : 16;
    const int lmb  = (lane < 16) ? 0 : 8;

    v8f acc[4][2];
    #pragma unroll
    for (int i = 0; i < 4; ++i)
        #pragma unroll
        for (int j = 0; j < 2; ++j) acc[i][j] = {};

    for (int kk = 0; kk < CC / 32; ++kk) {
        const int k0 = kk * 32;
        __syncthreads();
        #pragma unroll
        for (int j = 0; j < 4; ++j) {
            const int g   = tid + j * 256;
            const int row = g >> 3;
            const int c4  = (g & 7) * 4;
            float4 wv = *reinterpret_cast<const float4*>(W + (size_t)(o0 + row) * CC + k0 + c4);
            Ws[row * KP + c4 + 0] = f2bf(wv.x);
            Ws[row * KP + c4 + 1] = f2bf(wv.y);
            Ws[row * KP + c4 + 2] = f2bf(wv.z);
            Ws[row * KP + c4 + 3] = f2bf(wv.w);
        }
        __syncthreads();

        v16bf afr[4], bfr[2];
        #pragma unroll
        for (int ms = 0; ms < 4; ++ms) {
            const unsigned short* p = A + (size_t)(m0 + wm + ms * 16 + lrow) * CC + k0 + c0;
            afr[ms] = mkfrag(*(const v8u16*)p, *(const v8u16*)(p + 16));
        }
        #pragma unroll
        for (int os = 0; os < 2; ++os) {
            const unsigned short* p = &Ws[(wo + os * 16 + lrow) * KP + c0];
            bfr[os] = mkfrag(*(const v8u16*)p, *(const v8u16*)(p + 16));
        }
        #pragma unroll
        for (int ms = 0; ms < 4; ++ms)
            #pragma unroll
            for (int os = 0; os < 2; ++os)
                acc[ms][os] = wmma_bf16(afr[ms], bfr[os], acc[ms][os]);
    }

    #pragma unroll
    for (int ms = 0; ms < 4; ++ms)
        #pragma unroll
        for (int os = 0; os < 2; ++os) {
            const int o = o0 + wo + os * 16 + lrow;
            const float bv = bias[o];
            #pragma unroll
            for (int j = 0; j < 8; ++j) {
                const size_t m = (size_t)(m0 + wm + ms * 16 + lmb + j);
                out[m * CC + o] = acc[ms][os][j] + bv;
            }
        }
}

// ---------------------------------------------------------------------------
extern "C" void kernel_launch(void* const* d_in, const int* in_sizes, int n_in,
                              void* d_out, int out_size, void* d_ws, size_t ws_size,
                              hipStream_t stream) {
    (void)in_sizes; (void)n_in; (void)out_size; (void)ws_size;
    const float* x  = (const float*)d_in[0];
    const float* y  = (const float*)d_in[1];
    const float* Wq = (const float*)d_in[2];
    const float* Wk = (const float*)d_in[3];
    const float* Wv = (const float*)d_in[4];
    const float* Wp = (const float*)d_in[5];
    const float* bp = (const float*)d_in[6];
    float* out = (float*)d_out;

    const size_t CHUNK = (size_t)BB * NN * CC;  // 33,554,432 bf16 elems = 64 MB
    unsigned short* ws = (unsigned short*)d_ws;
    unsigned short* Qt = ws;               // [b][h][d][n] bf16
    unsigned short* Kt = ws + CHUNK;       // [b][h][e][n] bf16
    unsigned short* Vn = ws + 2 * CHUNK;   // (B*N, C) bf16
    unsigned short* Ot = ws + 3 * CHUNK;   // (B*N, C) bf16

    dim3 gG(BB * NN / 128, CC / 128);      // (256, 8)
    dim3 blk(256);

    qkv_gemm_kernel<<<gG, blk, 0, stream>>>(x, Wq, Qt, 1);
    qkv_gemm_kernel<<<gG, blk, 0, stream>>>(y, Wk, Kt, 1);
    qkv_gemm_kernel<<<gG, blk, 0, stream>>>(y, Wv, Vn, 0);
    attn_kernel<<<dim3(BB * HH), blk, 0, stream>>>(Qt, Kt, Vn, Ot);
    proj_kernel<<<gG, blk, 0, stream>>>(Ot, Wp, bp, out);
}
